// ToM2C_multi_22582938042883
// MI455X (gfx1250) — compile-verified, matrix operand order checked by reading, not run on hardware
//
#include <hip/hip_runtime.h>
#include <hip/hip_bf16.h>
#include <math.h>

#define BB   4
#define NND  192
#define HD   192
#define DIN  64
#define NT   12      // 192 / 16 N-tiles
#define KTS  6       // 192 / 32 K-steps
#define WAVES 8
#define TAUV 0.5f
#define WE   (HD * HD)   // 36864 f16 elements per heavy weight

typedef __attribute__((ext_vector_type(16))) _Float16 v16h;
typedef __attribute__((ext_vector_type(8)))  _Float16 v8h;
typedef __attribute__((ext_vector_type(8)))  float    v8f;
typedef __attribute__((ext_vector_type(4)))  unsigned int u32x4;
typedef __attribute__((ext_vector_type(8)))  int i32x8;
typedef __attribute__((ext_vector_type(4)))  int i32x4;

// ---------------- WMMA helpers ----------------------------------------------

__device__ __forceinline__ v8f wmma_f16(v16h a, v16h b, v8f c) {
  return __builtin_amdgcn_wmma_f32_16x16x32_f16(
      /*neg_a=*/false, a, /*neg_b=*/false, b,
      /*c_mod=*/(short)0, c, /*reuse_a=*/false, /*reuse_b=*/false);
}

// Load a 16x32 f16 fragment from a row-major tile (global or LDS).
// ISA layout: lane&15 = row; elems 0..7 <-> K+kh+0..7, 8..15 <-> K+kh+16..23.
__device__ __forceinline__ v16h load_frag16(const _Float16* base, int ldh, int lane) {
  int r  = lane & 15;
  int kh = (lane >> 4) << 3;
  const _Float16* p = base + r * ldh + kh;
  v8h lo = *(const v8h*)(p);
  v8h hi = *(const v8h*)(p + 16);
  v16h f;
#pragma unroll
  for (int t = 0; t < 8; ++t) { f[t] = lo[t]; f[t + 8] = hi[t]; }
  return f;
}

// B-fragment load from the fragment-ordered LDS weight image:
// frag (kt,nt) -> 1024 contiguous bytes; lane's 32 bytes at +lane*32.
__device__ __forceinline__ v16h load_frag_lin(const _Float16* wlane, int fragid) {
  const _Float16* p = wlane + (fragid << 9);
  v8h lo = *(const v8h*)(p);
  v8h hi = *(const v8h*)(p + 8);
  v16h f;
#pragma unroll
  for (int t = 0; t < 8; ++t) { f[t] = lo[t]; f[t + 8] = hi[t]; }
  return f;
}

// ---------------- Tensor Data Mover: linear 73,728 B weight -> LDS ----------
// D# per CDNA5 ISA: data_size=2B, 1-D 36864-elem tile.
__device__ __forceinline__ void tdm_load_w(const _Float16* lds_dst,
                                           const _Float16* gsrc) {
  unsigned lds = (unsigned)(uintptr_t)(const void*)lds_dst;
  unsigned long long ga = (unsigned long long)(uintptr_t)(const void*)gsrc;
  u32x4 g0;
  g0[0] = 1u;                                    // count=1 (valid user D#)
  g0[1] = lds;                                   // lds_addr (bytes)
  g0[2] = (unsigned)ga;                          // global_addr[31:0]
  g0[3] = ((unsigned)(ga >> 32) & 0x01FFFFFFu) | 0x80000000u;  // addr hi | type=2
  i32x8 g1;
  g1[0] = 0x00010000;                            // data_size=1 -> 2 bytes
  g1[1] = (int)((WE & 0xFFFFu) << 16);           // tensor_dim0[15:0]
  g1[2] = (int)(((WE >> 16) & 0xFFFFu) | (1u << 16)); // dim0[31:16], tensor_dim1=1
  g1[3] = (int)((WE & 0xFFFFu) << 16);           // tile_dim0 = 36864
  g1[4] = 1;                                     // tile_dim1 = 1, tile_dim2 = 0
  g1[5] = (int)WE;                               // tensor_dim0_stride
  g1[6] = 0;
  g1[7] = 0;
  i32x4 z4 = {0, 0, 0, 0};                       // groups 2/3 unused (<=2D)
  i32x8 z8 = {0, 0, 0, 0, 0, 0, 0, 0};
  __builtin_amdgcn_tensor_load_to_lds(g0, g1, z4, z4, z8, 0);
}

// ---------------- tiny utility kernels --------------------------------------

__global__ void k_zero(float* __restrict__ p, int n) {
  int i = blockIdx.x * blockDim.x + threadIdx.x;
  if (i < n) p[i] = 0.f;
}

__global__ void k_copy(float* __restrict__ d, const float* __restrict__ s, int n) {
  int i = blockIdx.x * blockDim.x + threadIdx.x;
  if (i < n) d[i] = s[i];
}

// W[k][n] (192x192 f32) -> f16 image in FRAGMENT ORDER:
// dst d: fragid=d>>9 (kt=fragid/12, nt=fragid%12); rem=d&511; lane=rem>>4;
// chunk c=(rem>>3)&1; t=d&7;  value = W[k][n] with
// n = nt*16 + (lane&15), k = kt*32 + (lane>>4)*8 + c*16 + t.
__global__ void k_convwt(const float* __restrict__ W, _Float16* __restrict__ Wt) {
  int d = blockIdx.x * blockDim.x + threadIdx.x;
  if (d >= WE) return;
  int fragid = d >> 9;
  int kt = fragid / NT, nt = fragid % NT;
  int rem = d & 511;
  int lane = rem >> 4;
  int c = (rem >> 3) & 1;
  int t = d & 7;
  int n = nt * 16 + (lane & 15);
  int k = kt * 32 + ((lane >> 4) << 3) + c * 16 + t;
  Wt[d] = (_Float16)W[k * HD + n];
}

// Generic small f32 GEMM: C = [C +] A@W [+bias], optional relu. One thread/elem.
__global__ void k_sgemm(const float* __restrict__ A, const float* __restrict__ W,
                        const float* __restrict__ bias, float* __restrict__ C,
                        int M, int K, int Nn, int relu, int acc) {
  int idx = blockIdx.x * blockDim.x + threadIdx.x;
  if (idx >= M * Nn) return;
  int m = idx / Nn, n = idx % Nn;
  float s = bias ? bias[n] : 0.f;
  if (acc) s += C[idx];
  const float* a = A + (size_t)m * K;
#pragma unroll 4
  for (int k = 0; k < K; ++k) s += a[k] * W[(size_t)k * Nn + n];
  if (relu) s = fmaxf(s, 0.f);
  C[idx] = s;
}

// edge_effect0 = edge_enc = relu(xr[b,i,:] + xs[b,j,:] + ee_b) stored f16
__global__ void k_init_edge(const float* __restrict__ xr, const float* __restrict__ xs,
                            const float* __restrict__ eeb, _Float16* __restrict__ eff) {
  unsigned idx = blockIdx.x * blockDim.x + threadIdx.x;
  if (idx >= (unsigned)BB * NND * NND * HD) return;
  int h = (int)(idx % HD);
  unsigned t = idx / HD;
  int j = (int)(t % NND); t /= NND;
  int i = (int)(t % NND);
  int b = (int)(t / NND);
  float v = xr[(b * NND + i) * HD + h] + xs[(b * NND + j) * HD + h] + eeb[h];
  eff[idx] = (_Float16)fmaxf(v, 0.f);
}

// ---------------- fused edge-propagator (the heavy kernel) ------------------
__global__ __launch_bounds__(256) void k_edge_prop(
    _Float16* __restrict__ eff, const float* __restrict__ rpart,
    const float* __restrict__ spart, const float* __restrict__ b1,
    const _Float16* __restrict__ w1t, const _Float16* __restrict__ w2t,
    const float* __restrict__ b2, float* __restrict__ agg) {
  __shared__ __align__(16) _Float16 sW1[WE];
  __shared__ __align__(16) _Float16 sW2[WE];
  __shared__ __align__(16) _Float16 hbuf[WAVES][16][200];

  int wave = threadIdx.x >> 5;
  int lane = threadIdx.x & 31;

  if (threadIdx.x < 32) {          // one wave drives the TDM
    tdm_load_w(sW1, w1t);
    tdm_load_w(sW2, w2t);
    __builtin_amdgcn_s_wait_tensorcnt((short)0);
  }
  __syncthreads();

  int tile = blockIdx.x * WAVES + wave;
  int row0 = tile << 4;
  int j0 = (tile % NT) << 4;       // 12 row-tiles per (b,i)
  int bi = tile / NT;              // = b*192 + i
  int b = bi / NND;
  int i = bi % NND;

  const _Float16* arow = eff + (unsigned)(row0 * HD);
  const _Float16* w1l = sW1 + (lane << 4);  // per-lane frag base
  const _Float16* w2l = sW2 + (lane << 4);
  int nlo = lane & 15;
  int mhi = (lane >> 4) << 3;

  // ---- GEMM1: eff @ W1c, B from LDS (fragment-ordered) ----
  v8f acc[NT] = {};
#pragma unroll 1
  for (int kt = 0; kt < KTS; ++kt) {
    v16h a = load_frag16(arow + kt * 32, HD, lane);
#pragma unroll
    for (int nt = 0; nt < NT; ++nt)
      acc[nt] = wmma_f16(a, load_frag_lin(w1l, kt * NT + nt), acc[nt]);
  }
  // ---- epilogue 1: + rpart + spart + b1, relu, park h in LDS (f16) ----
  const float* rp = rpart + bi * HD;
#pragma unroll
  for (int nt = 0; nt < NT; ++nt) {
    int col = nt * 16 + nlo;
    float rcb = rp[col] + b1[col];
#pragma unroll
    for (int v = 0; v < 8; ++v) {
      int m = mhi + v;
      float sv = spart[(b * NND + (j0 + m)) * HD + col];
      hbuf[wave][m][col] = (_Float16)fmaxf(acc[nt][v] + rcb + sv, 0.f);
    }
  }
  // ---- GEMM2: h @ W2 (A from LDS h-tile, B from LDS) ----
  v8f acc2[NT] = {};
#pragma unroll 1
  for (int kt = 0; kt < KTS; ++kt) {
    v16h a = load_frag16(&hbuf[wave][0][0] + kt * 32, 200, lane);
#pragma unroll
    for (int nt = 0; nt < NT; ++nt)
      acc2[nt] = wmma_f16(a, load_frag_lin(w2l, kt * NT + nt), acc2[nt]);
  }
  // ---- epilogue 2: bias+relu, f16 store (coalesced over lanes), agg atomics --
  _Float16* orow = eff + (unsigned)(row0 * HD);
  float* aggp = agg + bi * HD;
#pragma unroll
  for (int nt = 0; nt < NT; ++nt) {
    int col = nt * 16 + nlo;
    float bb = b2[col];
    float ssum = 0.f;
#pragma unroll
    for (int v = 0; v < 8; ++v) {
      float o = fmaxf(acc2[nt][v] + bb, 0.f);
      ssum += o;
      orow[(mhi + v) * HD + col] = (_Float16)o;
    }
    atomicAdd(&aggp[col], ssum);
  }
}

// ---------------- predictor + gumbel-softmax(hard) ---------------------------

__device__ __forceinline__ float gumbel_noise(int row, int e) {
  unsigned x = (unsigned)(row * 2 + e) + 42u * 0x9E3779B9u;
  x ^= x >> 16; x *= 0x7FEB352Du;
  x ^= x >> 15; x *= 0x846CA68Bu;
  x ^= x >> 16;
  float u = ((x >> 8) + 0.5f) * (1.0f / 16777216.0f);
  return -logf(-logf(u));
}

__global__ __launch_bounds__(256) void k_predict(
    const _Float16* __restrict__ eff, const float* __restrict__ xr,
    const float* __restrict__ xs, const float* __restrict__ eeb,
    const _Float16* __restrict__ prAt, const _Float16* __restrict__ prBt,
    const float* __restrict__ pr1b, const float* __restrict__ pr2w,
    const float* __restrict__ pr2b, float* __restrict__ out_logits,
    float* __restrict__ out_edge) {
  __shared__ __align__(16) _Float16 sWA[WE];
  __shared__ __align__(16) _Float16 sWB[WE];
  __shared__ __align__(16) _Float16 phb[WAVES][16][200];

  int wave = threadIdx.x >> 5;
  int lane = threadIdx.x & 31;

  if (threadIdx.x < 32) {
    tdm_load_w(sWA, prAt);
    tdm_load_w(sWB, prBt);
    __builtin_amdgcn_s_wait_tensorcnt((short)0);
  }
  __syncthreads();

  int tile = blockIdx.x * WAVES + wave;
  int row0 = tile << 4;
  int j0 = (tile % NT) << 4;
  int bi = tile / NT;
  int b = bi / NND;
  int i = bi % NND;

  // stage the (recomputed, never-materialized) edge_enc tile into LDS
  const float* xrp = xr + bi * HD;
  for (int idx = lane; idx < 16 * HD; idx += 32) {
    int m = idx / HD, k = idx % HD;
    float v = xrp[k] + xs[(b * NND + (j0 + m)) * HD + k] + eeb[k];
    phb[wave][m][k] = (_Float16)fmaxf(v, 0.f);
  }

  const _Float16* arow = eff + (unsigned)(row0 * HD);
  const _Float16* wAl = sWA + (lane << 4);
  const _Float16* wBl = sWB + (lane << 4);
  int nlo = lane & 15;
  int mhi = (lane >> 4) << 3;

  v8f acc[NT] = {};
#pragma unroll 1
  for (int kt = 0; kt < KTS; ++kt) {
    v16h a1 = load_frag16(arow + kt * 32, HD, lane);
    v16h a2 = load_frag16(&phb[wave][0][0] + kt * 32, 200, lane);
#pragma unroll
    for (int nt = 0; nt < NT; ++nt) {
      acc[nt] = wmma_f16(a1, load_frag_lin(wAl, kt * NT + nt), acc[nt]);
      acc[nt] = wmma_f16(a2, load_frag_lin(wBl, kt * NT + nt), acc[nt]);
    }
  }
  // ph = relu(acc + pr1_b) -> LDS (overwrites consumed edge_enc tile)
#pragma unroll
  for (int nt = 0; nt < NT; ++nt) {
    int col = nt * 16 + nlo;
    float bb = pr1b[col];
#pragma unroll
    for (int v = 0; v < 8; ++v)
      phb[wave][mhi + v][col] = (_Float16)fmaxf(acc[nt][v] + bb, 0.f);
  }
  // [192,2] head: lane = (row r, type e)
  int r = lane & 15;
  int e = lane >> 4;
  float s = pr2b[e];
#pragma unroll 4
  for (int k = 0; k < HD; ++k) s += (float)phb[wave][r][k] * pr2w[k * 2 + e];
  int row = row0 + r;
  out_logits[row * 2 + e] = s;

  // gumbel-softmax (tau=0.5, hard): straight-through forward == one-hot
  float so = __shfl_xor(s, 16, 32);
  float le = (s + gumbel_noise(row, e)) / TAUV;
  float lo = (so + gumbel_noise(row, e ^ 1)) / TAUV;
  if (e == 1) {
    float val = (le > lo) ? 1.f : 0.f;
    if (i == (j0 + r)) val = 0.f;
    out_edge[row] = val;
  }
}

// ---------------- host-side orchestration ------------------------------------

extern "C" void kernel_launch(void* const* d_in, const int* in_sizes, int n_in,
                              void* d_out, int out_size, void* d_ws, size_t ws_size,
                              hipStream_t stream) {
  const float* x     = (const float*)d_in[0];
  const float* ne_w  = (const float*)d_in[1];
  const float* ne_b  = (const float*)d_in[2];
  const float* ee_w  = (const float*)d_in[3];
  const float* ee_b  = (const float*)d_in[4];
  const float* np1_w = (const float*)d_in[5];
  const float* np1_b = (const float*)d_in[6];
  const float* np2_w = (const float*)d_in[7];
  const float* np2_b = (const float*)d_in[8];
  const float* ep_a_w[2] = {(const float*)d_in[9],  (const float*)d_in[13]};
  const float* ep_a_b[2] = {(const float*)d_in[10], (const float*)d_in[14]};
  const float* ep_b_w[2] = {(const float*)d_in[11], (const float*)d_in[15]};
  const float* ep_b_b[2] = {(const float*)d_in[12], (const float*)d_in[16]};
  const float* pr1_w = (const float*)d_in[17];
  const float* pr1_b = (const float*)d_in[18];
  const float* pr2_w = (const float*)d_in[19];
  const float* pr2_b = (const float*)d_in[20];
  (void)in_sizes; (void)n_in; (void)out_size; (void)ws_size;

  char* ws = (char*)d_ws;
  size_t off = 0;
  auto carve = [&](size_t bytes) -> void* {
    void* p = ws + off;
    off = (off + bytes + 255) & ~(size_t)255;
    return p;
  };

  _Float16* eff = (_Float16*)carve((size_t)BB * NND * NND * HD * sizeof(_Float16));
  _Float16* wt[6];
  for (int q = 0; q < 6; ++q) wt[q] = (_Float16*)carve((size_t)WE * sizeof(_Float16));
  const size_t SB = (size_t)BB * NND * HD * sizeof(float);  // 589,824 B
  float* node_enc = (float*)carve(SB);
  float* node_eff = (float*)carve(SB);
  float* nh       = (float*)carve(SB);
  float* rpart    = (float*)carve(SB);
  float* spart    = (float*)carve(SB);
  float* agg      = (float*)carve(SB);
  float* xr       = (float*)carve(SB);
  float* xs       = (float*)carve(SB);

  const int THR = 256;
  const int MH = BB * NND * HD;                       // 147456
  dim3 gS((MH + THR - 1) / THR);
  dim3 gW((WE + THR - 1) / THR);
  const unsigned ETOT = (unsigned)BB * NND * NND * HD;
  dim3 gE((ETOT + THR - 1) / THR);
  const int TILES = BB * NND * NND / 16;              // 9216
  dim3 gP(TILES / WAVES);                             // 1152 blocks x 8 waves

  // f16 heavy weights -> fragment-ordered images (TDM copies them linearly)
  k_convwt<<<gW, THR, 0, stream>>>(ep_a_w[0] + 2 * HD * HD, wt[0]);
  k_convwt<<<gW, THR, 0, stream>>>(ep_b_w[0],               wt[1]);
  k_convwt<<<gW, THR, 0, stream>>>(ep_a_w[1] + 2 * HD * HD, wt[2]);
  k_convwt<<<gW, THR, 0, stream>>>(ep_b_w[1],               wt[3]);
  k_convwt<<<gW, THR, 0, stream>>>(pr1_w,                   wt[4]);
  k_convwt<<<gW, THR, 0, stream>>>(pr1_w + HD * HD,         wt[5]);

  // encoders (node level, tiny)
  k_sgemm<<<gS, THR, 0, stream>>>(x, ne_w, ne_b, node_enc, BB * NND, DIN, HD, 1, 0);
  k_sgemm<<<gS, THR, 0, stream>>>(x, ee_w,            nullptr, xr, BB * NND, DIN, HD, 0, 0);
  k_sgemm<<<gS, THR, 0, stream>>>(x, ee_w + DIN * HD, nullptr, xs, BB * NND, DIN, HD, 0, 0);
  k_copy<<<gS, THR, 0, stream>>>(node_eff, node_enc, MH);
  k_init_edge<<<gE, THR, 0, stream>>>(xr, xs, ee_b, eff);

  // two propagation steps
  for (int s = 0; s < 2; ++s) {
    k_sgemm<<<gS, THR, 0, stream>>>(node_eff, ep_a_w[s],           nullptr, rpart, BB * NND, HD, HD, 0, 0);
    k_sgemm<<<gS, THR, 0, stream>>>(node_eff, ep_a_w[s] + HD * HD, nullptr, spart, BB * NND, HD, HD, 0, 0);
    k_zero<<<gS, THR, 0, stream>>>(agg, MH);
    k_edge_prop<<<gP, THR, 0, stream>>>(eff, rpart, spart, ep_a_b[s], wt[2 * s], wt[2 * s + 1],
                                        ep_b_b[s], agg);
    // node propagator: nh = relu(node_enc@W + node_eff@W' + agg@W'' + b)
    k_sgemm<<<gS, THR, 0, stream>>>(node_enc, np1_w,               np1_b,   nh, BB * NND, HD, HD, 0, 0);
    k_sgemm<<<gS, THR, 0, stream>>>(node_eff, np1_w + HD * HD,     nullptr, nh, BB * NND, HD, HD, 0, 1);
    k_sgemm<<<gS, THR, 0, stream>>>(agg,      np1_w + 2 * HD * HD, nullptr, nh, BB * NND, HD, HD, 1, 1);
    k_sgemm<<<gS, THR, 0, stream>>>(nh, np2_w, np2_b, node_eff, BB * NND, HD, HD, 1, 0);
  }

  // predictor + gumbel-softmax hard + self-loop mask
  float* out_logits = (float*)d_out;
  float* out_edge   = out_logits + (size_t)BB * NND * NND * 2;
  k_predict<<<gP, THR, 0, stream>>>(eff, xr, xs, ee_b, wt[4], wt[5], pr1_b, pr2_w, pr2_b,
                                    out_logits, out_edge);
}
